// ShiftedWindowAttention_34737695490456
// MI455X (gfx1250) — matrix operand order repaired and checked by reading, not compiled, not run on hardware
//
#include <hip/hip_runtime.h>
#include <hip/hip_bf16.h>
#include <math.h>

// ---------------------------------------------------------------------------
// CDNA5 (gfx1250) Swin shifted-window attention, wave32 + v_wmma_f32_16x16x32_f16
// ---------------------------------------------------------------------------

typedef __attribute__((ext_vector_type(16))) _Float16 v16h;
typedef __attribute__((ext_vector_type(8)))  float    v8f;

__device__ __forceinline__ v8f wmma_f16(v16h a, v16h b, v8f c) {
  // D = A(16x32) * B(32x16) + C(16x16 f32)
  return __builtin_amdgcn_wmma_f32_16x16x32_f16(false, a, false, b, (short)0, c,
                                                false, false);
}

// A fragment (16x32 f16) from row-major [M][K] storage.
// ISA 7.12.2: lane<16 -> M=lane, K={0..7,16..23}; lane>=16 -> K={8..15,24..31}.
// Per lane: two contiguous 16-byte runs -> 2x (ds|global)_load_b128.
__device__ __forceinline__ v16h ldA(const _Float16* s, int row0, int k0, int ld) {
  const int lane = threadIdx.x & 31;
  const int m = lane & 15;
  const int kb = (lane < 16) ? 0 : 8;
  const _Float16* p = s + (row0 + m) * ld + k0 + kb;
  v16h a;
#pragma unroll
  for (int i = 0; i < 8; ++i) a[i] = p[i];
#pragma unroll
  for (int i = 0; i < 8; ++i) a[8 + i] = p[16 + i];
  return a;
}

// B fragment (32x16 f16) when the source holds B transposed, i.e. row-major
// [N][K]. lane<16 -> N=lane, K={0..7,16..23}; lane>=16 -> K={8..15,24..31}.
// Per lane: two contiguous 16-byte runs -> 2x (ds|global)_load_b128.
__device__ __forceinline__ v16h ldB_nk(const _Float16* s, int k0, int n0, int ld) {
  const int lane = threadIdx.x & 31;
  const int n = lane & 15;
  const int kb = (lane < 16) ? 0 : 8;
  const _Float16* p = s + (n0 + n) * ld + k0 + kb;
  v16h b;
#pragma unroll
  for (int i = 0; i < 8; ++i) b[i] = p[i];
#pragma unroll
  for (int i = 0; i < 8; ++i) b[8 + i] = p[16 + i];
  return b;
}

// ---------------------------------------------------------------------------
// Kernel 0: one-shot weight convert + transpose to f16 [N][K] (L2-resident)
// ---------------------------------------------------------------------------
__global__ __launch_bounds__(256) void convert_weights(
    const float* __restrict__ wq, const float* __restrict__ wp,
    _Float16* __restrict__ wq_t, _Float16* __restrict__ wp_t) {
  const int idx = blockIdx.x * 256 + threadIdx.x;
  if (idx < 768 * 256) {
    const int n = idx >> 8, k = idx & 255;
    wq_t[idx] = (_Float16)wq[k * 768 + n];       // wq_t[n][k]
  } else {
    const int j = idx - 768 * 256;
    const int n = j >> 8, k = j & 255;
    wp_t[j] = (_Float16)wp[k * 256 + n];         // wp_t[n][k]
  }
}

// ---------------------------------------------------------------------------
// Kernel 1: fused shift-roll + window partition + QKV GEMM
//           M=49(pad 64), K=256, N=768. B-fragments straight from L2.
// ---------------------------------------------------------------------------
#define QKV_AS 264   // A tile stride (halfs): 528B rows, 16B-aligned chunks

__global__ __launch_bounds__(256) void swin_qkv(
    const float* __restrict__ x, const _Float16* __restrict__ wq_t,
    const float* __restrict__ bq, _Float16* __restrict__ q_ws,
    _Float16* __restrict__ k_ws, _Float16* __restrict__ v_ws) {
  __shared__ _Float16 As[64 * QKV_AS];
  const int bn = blockIdx.x;
  const int b = bn >> 6, w64 = bn & 63, wy = w64 >> 3, wx = w64 & 7;
  const int tid = threadIdx.x, wave = tid >> 5, lane = tid & 31;

  // Stage shifted-window activations (roll -3,-3 fused), f32x4 -> f16x4
  for (int i = tid; i < 64 * 64; i += 256) {
    const int row = i >> 6, col = (i & 63) << 2;
    float4 vv;
    vv.x = vv.y = vv.z = vv.w = 0.f;
    if (row < 49) {
      const int ty = row / 7, tx = row % 7;
      const int oy = (wy * 7 + ty + 3) % 56;
      const int ox = (wx * 7 + tx + 3) % 56;
      vv = *(const float4*)(x + ((b * 56 + oy) * 56 + ox) * 256 + col);
    }
    _Float16 h[4] = {(_Float16)vv.x, (_Float16)vv.y, (_Float16)vv.z,
                     (_Float16)vv.w};
    *(uint2*)&As[row * QKV_AS + col] = *(const uint2*)h;
  }
  __syncthreads();

  for (int half = 0; half < 2; ++half) {  // N = 768 in two 384 halves
    v8f acc[4][3];
#pragma unroll
    for (int mt = 0; mt < 4; ++mt)
#pragma unroll
      for (int j = 0; j < 3; ++j)
#pragma unroll
        for (int e = 0; e < 8; ++e) acc[mt][j][e] = 0.f;

    for (int ks = 0; ks < 8; ++ks) {  // K = 256 in 8 steps of 32
      const int k0 = ks * 32;
      v16h bf[3];
#pragma unroll
      for (int j = 0; j < 3; ++j)
        bf[j] = ldB_nk(wq_t, k0, half * 384 + wave * 48 + j * 16, 256);
#pragma unroll
      for (int mt = 0; mt < 4; ++mt) {
        v16h a = ldA(As, mt * 16, k0, QKV_AS);
#pragma unroll
        for (int j = 0; j < 3; ++j) acc[mt][j] = wmma_f16(a, bf[j], acc[mt][j]);
      }
    }

    // Epilogue: +bias (q pre-scaled by hd^-0.5), split q/k/v [Bn][h][49][32]
#pragma unroll
    for (int mt = 0; mt < 4; ++mt)
#pragma unroll
      for (int j = 0; j < 3; ++j) {
        const int gn = half * 384 + wave * 48 + j * 16 + (lane & 15);
        const int part = gn >> 8, head = (gn >> 5) & 7, d = gn & 31;
        _Float16* dst = (part == 0) ? q_ws : (part == 1) ? k_ws : v_ws;
        const float bias = bq[gn];
        const float scale = (part == 0) ? 0.17677669529663689f : 1.0f;
#pragma unroll
        for (int r = 0; r < 8; ++r) {
          const int m = mt * 16 + r + ((lane < 16) ? 0 : 8);
          if (m < 49)
            dst[((bn * 8 + head) * 49 + m) * 32 + d] =
                (_Float16)((acc[mt][j][r] + bias) * scale);
        }
      }
  }
}

// ---------------------------------------------------------------------------
// Kernel 2: per (window, head) attention, one wave32 per block.
//   S = q k^T + rel_bias + shift_mask ; softmax ; O = S v
// ---------------------------------------------------------------------------
#define ATT_QS 40   // q/k LDS stride (32 + skew), 80B rows
#define ATT_VS 72   // v^T LDS stride (64 + skew), 144B rows
#define ATT_FS 66   // f32 attn stride
#define ATT_HS 72   // f16 attn stride

__global__ __launch_bounds__(32) void swin_attn(
    const _Float16* __restrict__ q_ws, const _Float16* __restrict__ k_ws,
    const _Float16* __restrict__ v_ws, const float* __restrict__ table,
    _Float16* __restrict__ o_ws) {
  __shared__ _Float16 qs[64 * ATT_QS], ks[64 * ATT_QS];
  __shared__ _Float16 vsT[32 * ATT_VS];          // v transposed: [d][token]
  __shared__ float af[64 * ATT_FS];
  __shared__ _Float16 a16[64 * ATT_HS];
  const int bn = blockIdx.x >> 3, head = blockIdx.x & 7;
  const int lane = threadIdx.x;
  const int base = ((bn * 8 + head) * 49) * 32;

  // Vector-stage q,k (row-major) and v (transposed), zero-pad rows 49..63
  for (int c = lane; c < 256; c += 32) {      // 64 rows x 4 chunks of 8 halfs
    const int row = c >> 2, colc = (c & 3) << 3;
    uint4 z;
    z.x = z.y = z.z = z.w = 0u;
    uint4 qv = z, kv = z, vv = z;
    if (row < 49) {
      qv = *(const uint4*)(q_ws + base + row * 32 + colc);
      kv = *(const uint4*)(k_ws + base + row * 32 + colc);
      vv = *(const uint4*)(v_ws + base + row * 32 + colc);
    }
    *(uint4*)&qs[row * ATT_QS + colc] = qv;
    *(uint4*)&ks[row * ATT_QS + colc] = kv;
    _Float16 vh[8];
    *(uint4*)vh = vv;
#pragma unroll
    for (int e = 0; e < 8; ++e) vsT[(colc + e) * ATT_VS + row] = vh[e];
  }
  __syncthreads();

  // S = q k^T : 4x4 tiles, single K=32 WMMA step (k as [N][K] -> ldB_nk)
  v16h kb[4];
#pragma unroll
  for (int nt = 0; nt < 4; ++nt) kb[nt] = ldB_nk(ks, 0, nt * 16, ATT_QS);
  v8f acc[4][4];
#pragma unroll
  for (int mt = 0; mt < 4; ++mt) {
    v16h a = ldA(qs, mt * 16, 0, ATT_QS);
#pragma unroll
    for (int nt = 0; nt < 4; ++nt) {
#pragma unroll
      for (int e = 0; e < 8; ++e) acc[mt][nt][e] = 0.f;
      acc[mt][nt] = wmma_f16(a, kb[nt], acc[mt][nt]);
    }
  }

  // + relative position bias + shifted-window region mask, spill to LDS f32
  const int w64 = bn & 63, wy = w64 >> 3, wx = w64 & 7;
#pragma unroll
  for (int mt = 0; mt < 4; ++mt)
#pragma unroll
    for (int nt = 0; nt < 4; ++nt)
#pragma unroll
      for (int r = 0; r < 8; ++r) {
        const int m = mt * 16 + r + ((lane < 16) ? 0 : 8);
        const int n = nt * 16 + (lane & 15);
        float v = acc[mt][nt][r];
        if (m < 49 && n < 49) {
          const int iy = m / 7, ix = m % 7, jy = n / 7, jx = n % 7;
          v += table[((iy - jy + 6) * 13 + (ix - jx + 6)) * 8 + head];
          const int gyi = wy * 7 + iy, gxi = wx * 7 + ix;
          const int gyj = wy * 7 + jy, gxj = wx * 7 + jx;
          const int ci = (gyi < 49 ? 0 : (gyi < 53 ? 1 : 2)) * 3 +
                         (gxi < 49 ? 0 : (gxi < 53 ? 1 : 2));
          const int cj = (gyj < 49 ? 0 : (gyj < 53 ? 1 : 2)) * 3 +
                         (gxj < 49 ? 0 : (gxj < 53 ? 1 : 2));
          if (ci != cj) v -= 100.f;
        }
        af[m * ATT_FS + n] = v;
      }
  __syncthreads();

  // Row softmax over 49 valid keys (2 rows per lane), emit f16 attn (pad 0)
#pragma unroll
  for (int rr = 0; rr < 2; ++rr) {
    const int row = lane + rr * 32;
    if (row < 49) {
      float mx = -3.0e38f;
      for (int n = 0; n < 49; ++n) mx = fmaxf(mx, af[row * ATT_FS + n]);
      float sum = 0.f;
      for (int n = 0; n < 49; ++n) sum += __expf(af[row * ATT_FS + n] - mx);
      const float inv = 1.f / sum;
      for (int n = 0; n < 64; ++n)
        a16[row * ATT_HS + n] =
            (n < 49) ? (_Float16)(__expf(af[row * ATT_FS + n] - mx) * inv)
                     : (_Float16)0.f;
    } else {
      for (int n = 0; n < 64; ++n) a16[row * ATT_HS + n] = (_Float16)0.f;
    }
  }
  __syncthreads();

  // O = S v : M=64, N=32, K=64 (two K=32 steps; pad rows are zero)
  v8f oacc[4][2];
#pragma unroll
  for (int mt = 0; mt < 4; ++mt)
#pragma unroll
    for (int nt = 0; nt < 2; ++nt)
#pragma unroll
      for (int e = 0; e < 8; ++e) oacc[mt][nt][e] = 0.f;
#pragma unroll
  for (int kt = 0; kt < 2; ++kt) {
    v16h b0 = ldB_nk(vsT, kt * 32, 0, ATT_VS);
    v16h b1 = ldB_nk(vsT, kt * 32, 16, ATT_VS);
#pragma unroll
    for (int mt = 0; mt < 4; ++mt) {
      v16h a = ldA(a16, mt * 16, kt * 32, ATT_HS);
      oacc[mt][0] = wmma_f16(a, b0, oacc[mt][0]);
      oacc[mt][1] = wmma_f16(a, b1, oacc[mt][1]);
    }
  }

  // Concatenate heads: o_ws[bn][token][head*32+d] f16
#pragma unroll
  for (int mt = 0; mt < 4; ++mt)
#pragma unroll
    for (int nt = 0; nt < 2; ++nt)
#pragma unroll
      for (int r = 0; r < 8; ++r) {
        const int m = mt * 16 + r + ((lane < 16) ? 0 : 8);
        if (m < 49) {
          const int d = nt * 16 + (lane & 15);
          o_ws[(bn * 49 + m) * 256 + head * 32 + d] = (_Float16)oacc[mt][nt][r];
        }
      }
}

// ---------------------------------------------------------------------------
// Kernel 3: output projection GEMM + fused window-reverse + roll-back scatter
// ---------------------------------------------------------------------------
#define PRJ_S 264

__global__ __launch_bounds__(256) void swin_proj(
    const _Float16* __restrict__ o_ws, const _Float16* __restrict__ wp_t,
    const float* __restrict__ bp, float* __restrict__ out) {
  __shared__ _Float16 As[64 * PRJ_S];
  const int bn = blockIdx.x;
  const int b = bn >> 6, w64 = bn & 63, wy = w64 >> 3, wx = w64 & 7;
  const int tid = threadIdx.x, wave = tid >> 5, lane = tid & 31;

  // Vector-stage activations (f16, raw 16B copies), zero-pad rows 49..63
  for (int c = tid; c < 64 * 32; c += 256) {  // 64 rows x 32 chunks of 8 halfs
    const int row = c >> 5, colc = (c & 31) << 3;
    uint4 vv;
    vv.x = vv.y = vv.z = vv.w = 0u;
    if (row < 49)
      vv = *(const uint4*)(o_ws + (bn * 49 + row) * 256 + colc);
    *(uint4*)&As[row * PRJ_S + colc] = vv;
  }
  __syncthreads();

  v8f acc[4][2];
#pragma unroll
  for (int mt = 0; mt < 4; ++mt)
#pragma unroll
    for (int j = 0; j < 2; ++j)
#pragma unroll
      for (int e = 0; e < 8; ++e) acc[mt][j][e] = 0.f;

  for (int ks = 0; ks < 8; ++ks) {
    const int k0 = ks * 32;
    v16h bf0 = ldB_nk(wp_t, k0, wave * 32, 256);
    v16h bf1 = ldB_nk(wp_t, k0, wave * 32 + 16, 256);
#pragma unroll
    for (int mt = 0; mt < 4; ++mt) {
      v16h a = ldA(As, mt * 16, k0, PRJ_S);
      acc[mt][0] = wmma_f16(a, bf0, acc[mt][0]);
      acc[mt][1] = wmma_f16(a, bf1, acc[mt][1]);
    }
  }

  // +bias, window-reverse + roll(+3,+3) scatter into [B,56,56,256] fp32
#pragma unroll
  for (int mt = 0; mt < 4; ++mt)
#pragma unroll
    for (int j = 0; j < 2; ++j) {
      const int n = wave * 32 + j * 16 + (lane & 15);
      const float bias = bp[n];
#pragma unroll
      for (int r = 0; r < 8; ++r) {
        const int m = mt * 16 + r + ((lane < 16) ? 0 : 8);
        if (m < 49) {
          const int ty = m / 7, tx = m % 7;
          const int y = (wy * 7 + ty + 3) % 56;
          const int xg = (wx * 7 + tx + 3) % 56;
          out[((b * 56 + y) * 56 + xg) * 256 + n] = acc[mt][j][r] + bias;
        }
      }
    }
}

// ---------------------------------------------------------------------------
extern "C" void kernel_launch(void* const* d_in, const int* in_sizes, int n_in,
                              void* d_out, int out_size, void* d_ws,
                              size_t ws_size, hipStream_t stream) {
  const float* x  = (const float*)d_in[0];
  const float* qw = (const float*)d_in[1];
  const float* qb = (const float*)d_in[2];
  const float* pw = (const float*)d_in[3];
  const float* pb = (const float*)d_in[4];
  const float* bt = (const float*)d_in[5];

  // f16 workspace: q,k,v [2048][8][49][32]; attn-out [2048][49][256];
  // transposed f16 weights (L2-resident).
  const size_t P = (size_t)2048 * 8 * 49 * 32;  // 25,690,112 halfs each
  _Float16* q_ws = (_Float16*)d_ws;
  _Float16* k_ws = q_ws + P;
  _Float16* v_ws = k_ws + P;
  _Float16* o_ws = v_ws + P;
  _Float16* wq_t = o_ws + P;            // [768][256]
  _Float16* wp_t = wq_t + 768 * 256;    // [256][256]

  convert_weights<<<1024, 256, 0, stream>>>(qw, pw, wq_t, wp_t);
  swin_qkv<<<2048, 256, 0, stream>>>(x, wq_t, qb, q_ws, k_ws, v_ws);
  swin_attn<<<16384, 32, 0, stream>>>(q_ws, k_ws, v_ws, bt, o_ws);
  swin_proj<<<2048, 256, 0, stream>>>(o_ws, wp_t, pb, (float*)d_out);
}
